// AttentionDecoder_28561532518888
// MI455X (gfx1250) — matrix operand order repaired and checked by reading, not compiled, not run on hardware
//
#include <hip/hip_runtime.h>

// Problem constants (match reference)
#define B_ 32
#define S_ 4096
#define D_ 256
#define H_ 8
#define K_ 4
#define V_ 32
#define O_ 16
#define NSPLIT 32
#define SCHUNK (S_ / NSPLIT)   // 128 rows of S per block, 8 subtiles of 16

typedef __attribute__((ext_vector_type(16))) __bf16 v16bf;
typedef __attribute__((ext_vector_type(8)))  float  v8f;
typedef __attribute__((ext_vector_type(2)))  float  v2f;
typedef __attribute__((ext_vector_type(4)))  unsigned v4u;
typedef __attribute__((ext_vector_type(8)))  int    v8i;
typedef __attribute__((ext_vector_type(4)))  int    v4i;

__device__ __forceinline__ unsigned pack_bf16(float a, float b) {
  // round-to-nearest-even f32 -> bf16, packed pair (lo = a, hi = b)
  unsigned ua = __float_as_uint(a), ub = __float_as_uint(b);
  ua = (ua + 0x7fffu + ((ua >> 16) & 1u)) >> 16;
  ub = (ub + 0x7fffu + ((ub >> 16) & 1u)) >> 16;
  return (ua & 0xffffu) | (ub << 16);
}

// DPP row_ror:n (rotation within each 16-lane row) — pure VALU lane exchange.
#define DPP_ROR(x, n) \
  __int_as_float(__builtin_amdgcn_update_dpp(0, __float_as_int(x), 0x120 + (n), 0xf, 0xf, true))

__device__ __forceinline__ float redmax16(float v) {
  v = fmaxf(v, DPP_ROR(v, 1));
  v = fmaxf(v, DPP_ROR(v, 2));
  v = fmaxf(v, DPP_ROR(v, 4));
  v = fmaxf(v, DPP_ROR(v, 8));
  return v;
}
__device__ __forceinline__ float redsum16(float v) {
  v += DPP_ROR(v, 1);
  v += DPP_ROR(v, 2);
  v += DPP_ROR(v, 4);
  v += DPP_ROR(v, 8);
  return v;
}

// Dynamic-LDS layout (byte offsets, all 32B aligned)
#define LW_OFF   0        // 36864 u32 (147456B): W bf16 frags [nt=18][kt=8][lane=32][8 u32]
#define LA_OFF   147456   //  2048 u32 (  8192B): A bf16 frags [kt=8][lane=32][8 u32]
#define LQ_OFF   155648   //   512 f32: queries [O][H][K]
#define LBK_OFF  157696   //    32 f32: bk
#define LBV_OFF  157824   //   256 f32: bv
#define LKEY_OFF 158848   //   512 f32: keys tile [16 s][32]
#define LV_OFF   160896   //  4096 f32: values tile [8 head][16 s][32 v]
#define LP_OFF   177280   //  2048 f32: probs [8 head][16 s][16 o]
#define LMSK_OFF 185472   //    16 f32: mask tile
#define LRAW_OFF 185536   //  4096 f32 (16384B): TDM-staged raw input tile [16 s][256 d]
#define SMEM_BYTES 201920

__global__ __launch_bounds__(256, 1) void attn_fused_partial(
    const float* __restrict__ x, const int* __restrict__ msk,
    const float* __restrict__ Wk, const float* __restrict__ bk,
    const float* __restrict__ Wv, const float* __restrict__ bv,
    const float* __restrict__ q,
    float* __restrict__ wsM, float* __restrict__ wsL, float* __restrict__ wsA)
{
  extern __shared__ __align__(32) char smem[];
  unsigned* lw = (unsigned*)(smem + LW_OFF);
  unsigned* la = (unsigned*)(smem + LA_OFF);
  float* lq   = (float*)(smem + LQ_OFF);
  float* lbk  = (float*)(smem + LBK_OFF);
  float* lbv  = (float*)(smem + LBV_OFF);
  float* lke  = (float*)(smem + LKEY_OFF);
  float* lv   = (float*)(smem + LV_OFF);
  float* lp   = (float*)(smem + LP_OFF);
  float* lm   = (float*)(smem + LMSK_OFF);
  const float* lraw = (const float*)(smem + LRAW_OFF);

  const int tid  = threadIdx.x;
  const int wv   = tid >> 5;        // wave id == head id (8 waves, H=8)
  const int lane = tid & 31;
  const int ln   = lane & 15;
  const int hil  = lane >> 4;
  const int b    = blockIdx.x >> 5;
  const int sp   = blockIdx.x & 31;
  const int sbase = sp * SCHUNK;

  // ---- Stage combined W = [Wk | Wv] (D=256 x N=288) as bf16 in WMMA-B
  //      fragment order, this code's contiguous-k variant (lane = col, half selects K 0..15/16..31).
  for (int i = 0; i < 144; ++i) {
    int idx = tid + (i << 8);             // 0..36863
    int j8 = idx & 7;
    int l  = (idx >> 3) & 31;
    int kt = (idx >> 8) & 7;
    int nt = idx >> 11;
    int col = nt * 16 + (l & 15);
    int k0  = kt * 32 + ((l >> 4) << 4) + 2 * j8;
    float w0, w1;
    if (col < 32) { w0 = Wk[k0 * 32 + col];       w1 = Wk[(k0 + 1) * 32 + col]; }
    else { int c = col - 32; w0 = Wv[k0 * 256 + c]; w1 = Wv[(k0 + 1) * 256 + c]; }
    lw[idx] = pack_bf16(w0, w1);
  }
  for (int i = tid; i < O_ * H_ * K_; i += 256) lq[i] = q[i];
  if (tid < 32)  lbk[tid] = bk[tid];
  lbv[tid] = bv[tid];

  // Per-wave persistent flash-attention state for head `wv`:
  // rows o = r + 8*hil ; acc lanes are v columns.
  float mrun[8], lrun[8];
  v8f acc0 = {}, acc1 = {};
  #pragma unroll
  for (int r = 0; r < 8; ++r) { mrun[r] = -3.0e38f; lrun[r] = 0.0f; }
  const int ntv0 = 2 + 2 * wv;     // combined-N tile of this head's V cols 0..15
  const int ntv1 = 3 + 2 * wv;     // cols 16..31

  for (int st = 0; st < 8; ++st) {
    const int s0 = sbase + st * 16;
    __syncthreads();   // previous iter consumers of la/lke/lv/lp/lraw done

    // ---- TDM: DMA the contiguous 16x256 f32 input tile into LDS.
    //      1-D descriptor: data_size=4B, tile_dim0 = 4096 elements (16KB).
    if (wv == 0) {
      unsigned long long ga =
          (unsigned long long)(const void*)(x + ((size_t)b * S_ + s0) * D_);
      v4u g0;
      g0[0] = 1u;                                   // count=1, user-mode load
      g0[1] = (unsigned)LRAW_OFF;                   // lds_addr (bytes)
      g0[2] = (unsigned)(ga & 0xffffffffu);         // global_addr[31:0]
      g0[3] = (unsigned)((ga >> 32) & 0x01ffffffu)  // global_addr[56:32]
              | 0x80000000u;                        // type=2 ("image")
      v8i g1;
      g1[0] = 0x00020000;                           // data_size=4B; no mask/flags
      g1[1] = (int)(4096u << 16);                   // tensor_dim0[15:0] in [31:16]
      g1[2] = (int)(1u << 16);                      // dim0[31:16]=0 | tensor_dim1=1
      g1[3] = (int)(4096u << 16);                   // dim1 hi=0 | tile_dim0=4096
      g1[4] = 1;                                    // tile_dim1=1, tile_dim2=0
      g1[5] = 4096;                                 // tensor_dim0_stride lo32
      g1[6] = 0;
      g1[7] = 0;
      v4i z4 = {0, 0, 0, 0};
      v8i z8 = {0, 0, 0, 0, 0, 0, 0, 0};
      __builtin_amdgcn_tensor_load_to_lds(g0, g1, z4, z4, z8, 0);
      __builtin_amdgcn_s_wait_tensorcnt(0);
    }
    if (tid < 16) lm[tid] = (float)msk[b * S_ + s0 + tid];
    if (st < 7) {  // L2 prefetch of next subtile (global_prefetch_b8)
      const float* pn = x + (((size_t)b * S_ + (s0 + 16 + (tid >> 4))) * D_ + ((tid & 15) << 4));
      __builtin_prefetch(pn, 0, 3);
    }
    __syncthreads();   // lraw ready

    // ---- Convert staged f32 tile -> bf16 A fragments (16x32 A, contiguous k: half selects K 0..15/16..31, same as W)
    #pragma unroll
    for (int j = 0; j < 8; ++j) {
      int idx = tid + (j << 8);           // 0..2047
      int j8 = idx & 7;
      int l  = (idx >> 3) & 31;
      int kt = idx >> 8;
      int m  = l & 15;
      int kb = (l >> 4) << 4;
      int klo = kb + 2 * j8;
      int d = kt * 32 + klo;
      la[idx] = pack_bf16(lraw[m * D_ + d], lraw[m * D_ + d + 1]);
    }
    __syncthreads();   // la ready

    // ---- Projections with bf16 WMMA (f32 accumulate).
    v8f vc0 = {}, vc1 = {}, kc = {};
    #pragma unroll
    for (int kt = 0; kt < 8; ++kt) {
      v16bf a  = *(const v16bf*)(la + ((kt * 32 + lane) << 3));
      v16bf b0 = *(const v16bf*)(lw + (((ntv0 * 8 + kt) * 32 + lane) << 3));
      v16bf b1 = *(const v16bf*)(lw + (((ntv1 * 8 + kt) * 32 + lane) << 3));
      vc0 = __builtin_amdgcn_wmma_f32_16x16x32_bf16(false, a, false, b0, (short)0, vc0, false, false);
      vc1 = __builtin_amdgcn_wmma_f32_16x16x32_bf16(false, a, false, b1, (short)0, vc1, false, false);
      if ((wv & 3) == 0) {  // waves 0,4 also produce the two keys tiles
        v16bf bkf = *(const v16bf*)(lw + ((((wv >> 2) * 8 + kt) * 32 + lane) << 3));
        kc = __builtin_amdgcn_wmma_f32_16x16x32_bf16(false, a, false, bkf, (short)0, kc, false, false);
      }
    }
    // values (+bias) -> LDS [head][s][v]; C-layout: lane = v col, r -> s row
    #pragma unroll
    for (int r = 0; r < 8; ++r) {
      int srow = r + 8 * hil;
      lv[(wv * 16 + srow) * 32 + ln]      = vc0[r] + lbv[wv * 32 + ln];
      lv[(wv * 16 + srow) * 32 + 16 + ln] = vc1[r] + lbv[wv * 32 + 16 + ln];
    }
    if ((wv & 3) == 0) {  // keys (+bias, * 1/sqrt(K)) -> LDS [s][32]
      int cb = (wv >> 2) << 4;
      #pragma unroll
      for (int r = 0; r < 8; ++r) {
        int srow = r + 8 * hil;
        lke[srow * 32 + cb + ln] = 0.5f * (kc[r] + lbk[cb + ln]);
      }
    }
    __syncthreads();   // keys visible to all heads

    // ---- Scores: one f32 WMMA per head: D[16o x 16s] = Q[16o x 4] @ K^T[4 x 16s]
    v2f qa, kb2;
    qa.x  = lq[(ln * H_ + wv) * K_ + 2 * hil];
    qa.y  = lq[(ln * H_ + wv) * K_ + 2 * hil + 1];
    kb2.x = lke[ln * 32 + wv * 4 + 2 * hil];
    kb2.y = lke[ln * 32 + wv * 4 + 2 * hil + 1];
    v8f sc = {};
    sc = __builtin_amdgcn_wmma_f32_16x16x4_f32(false, qa, false, kb2, (short)0, sc, false, false);

    // ---- Online softmax over this 16-s tile (C-layout: lane = s, r+8*hil = o)
    //      16-lane reductions via DPP row rotations (VALU only, no LDS traffic).
    const float pen = 1.0e8f * (1.0f - lm[ln]);
    #pragma unroll
    for (int r = 0; r < 8; ++r) {
      float s = sc[r] - pen;
      float mn = fmaxf(mrun[r], redmax16(s));
      float al = __expf(mrun[r] - mn);
      float p  = __expf(s - mn);
      float ls = redsum16(p);
      lrun[r] = lrun[r] * al + ls;
      mrun[r] = mn;
      acc0[r] *= al;        // alpha is lane-uniform and row-matched to acc
      acc1[r] *= al;
      lp[(wv * 16 + ln) * 16 + r + 8 * hil] = p;   // [head][s][o]
    }
    // (lp/lv producers == consumers within the same wave; DS ops are in-order)

    // ---- acc[16o x 32v] += P[16o x 16s] @ V[16s x 32v] via f32 WMMA, 4 K-steps
    #pragma unroll
    for (int kk = 0; kk < 4; ++kk) {
      int sr = kk * 4 + 2 * hil;
      v2f pa, vb0, vb1;
      pa.x  = lp[(wv * 16 + sr)     * 16 + ln];
      pa.y  = lp[(wv * 16 + sr + 1) * 16 + ln];
      vb0.x = lv[(wv * 16 + sr)     * 32 + ln];
      vb0.y = lv[(wv * 16 + sr + 1) * 32 + ln];
      vb1.x = lv[(wv * 16 + sr)     * 32 + 16 + ln];
      vb1.y = lv[(wv * 16 + sr + 1) * 32 + 16 + ln];
      acc0 = __builtin_amdgcn_wmma_f32_16x16x4_f32(false, pa, false, vb0, (short)0, acc0, false, false);
      acc1 = __builtin_amdgcn_wmma_f32_16x16x4_f32(false, pa, false, vb1, (short)0, acc1, false, false);
    }
  }

  // ---- Write per-(b, split, head) flash partials
  const int obase = ((b * NSPLIT + sp) * H_ + wv) * O_;
  if (ln == 0) {
    #pragma unroll
    for (int r = 0; r < 8; ++r) {
      int o = r + 8 * hil;
      wsM[obase + o] = mrun[r];
      wsL[obase + o] = lrun[r];
    }
  }
  #pragma unroll
  for (int r = 0; r < 8; ++r) {
    int o = r + 8 * hil;
    wsA[(obase + o) * V_ + ln]      = acc0[r];
    wsA[(obase + o) * V_ + 16 + ln] = acc1[r];
  }
}

// Merge the NSPLIT partials: one wave per (b, h, o); 32 lanes == V columns.
__global__ __launch_bounds__(256) void attn_reduce(
    const float* __restrict__ wsM, const float* __restrict__ wsL,
    const float* __restrict__ wsA, float* __restrict__ out)
{
  int g = blockIdx.x * 8 + (threadIdx.x >> 5);
  int lane = threadIdx.x & 31;
  if (g >= B_ * H_ * O_) return;
  int b = g >> 7;
  int h = (g >> 4) & 7;
  int o = g & 15;
  float M = -3.0e38f;
  for (int i = 0; i < NSPLIT; ++i)
    M = fmaxf(M, wsM[((b * NSPLIT + i) * H_ + h) * O_ + o]);
  float L = 0.0f, a = 0.0f;
  for (int i = 0; i < NSPLIT; ++i) {
    int base = ((b * NSPLIT + i) * H_ + h) * O_ + o;
    float e = __expf(wsM[base] - M);
    L += wsL[base] * e;
    a += e * wsA[base * V_ + lane];
  }
  out[((size_t)b * O_ + o) * (H_ * V_) + h * V_ + lane] = a / L;
}

extern "C" void kernel_launch(void* const* d_in, const int* in_sizes, int n_in,
                              void* d_out, int out_size, void* d_ws, size_t ws_size,
                              hipStream_t stream) {
  const float* x  = (const float*)d_in[0];
  const int*   mk = (const int*)d_in[1];
  const float* Wk = (const float*)d_in[2];
  const float* bk = (const float*)d_in[3];
  const float* Wv = (const float*)d_in[4];
  const float* bv = (const float*)d_in[5];
  const float* q  = (const float*)d_in[6];

  const size_t nPart = (size_t)B_ * NSPLIT * H_ * O_;   // 131072
  float* wsM = (float*)d_ws;
  float* wsL = wsM + nPart;
  float* wsA = wsL + nPart;                             // nPart * V floats

  attn_fused_partial<<<B_ * NSPLIT, 256, SMEM_BYTES, stream>>>(
      x, mk, Wk, bk, Wv, bv, q, wsM, wsL, wsA);
  attn_reduce<<<(B_ * H_ * O_) / 8, 256, 0, stream>>>(
      wsM, wsL, wsA, (float*)d_out);
}
